// WeiAttention_5746666242529
// MI455X (gfx1250) — compile-verified, matrix operand order checked by reading, not run on hardware
//
#include <hip/hip_runtime.h>
#include <hip/hip_bf16.h>

// ---------------------------------------------------------------------------
// Flash-attention-2 forward, B=4,H=16,S=2048,D=64 (fp32 io), causal.
// CDNA5 (gfx1250) wave32, V_WMMA_F32_16X16X32_F16.
// Block = 8 waves = 128 query rows; 64-key blocks cooperatively staged into
// double-buffered LDS tiles as f16 (V transposed, key axis permuted to match
// packed-P storage). One barrier per key block; B-operand registers are
// double-buffered so ds_load latency hides under WMMA execution.
// ---------------------------------------------------------------------------

typedef _Float16 v16h __attribute__((ext_vector_type(16)));
typedef _Float16 v8h  __attribute__((ext_vector_type(8)));
typedef _Float16 v2h  __attribute__((ext_vector_type(2)));
typedef float    v8f  __attribute__((ext_vector_type(8)));
typedef float    v4f  __attribute__((ext_vector_type(4)));

#define WMMA_F16(a, b, c) \
    __builtin_amdgcn_wmma_f32_16x16x32_f16(false, (a), false, (b), (short)0, (c), false, false)

// ds_swizzle_b32 XOR exchange within 16-lane halves (group-of-32 mode).
template <int PAT>
static __device__ __forceinline__ float swz(float x) {
    return __int_as_float(__builtin_amdgcn_ds_swizzle(__float_as_int(x), PAT));
}
static __device__ __forceinline__ float rowmax16(float x) {
    x = fmaxf(x, swz<0x041F>(x));
    x = fmaxf(x, swz<0x081F>(x));
    x = fmaxf(x, swz<0x101F>(x));
    x = fmaxf(x, swz<0x201F>(x));
    return x;
}
static __device__ __forceinline__ float rowsum16(float x) {
    x += swz<0x041F>(x);
    x += swz<0x081F>(x);
    x += swz<0x101F>(x);
    x += swz<0x201F>(x);
    return x;
}

// 16-bit WMMA operand (ISA 7.12.2): elements 0..7 -> K=kb..kb+7,
// 8..15 -> K=kb+16..kb+23. Union load -> two ds_load_b128, no register moves.
union Op32 { v16h v; v8h h[2]; };
static __device__ __forceinline__ v16h lds_operand(const _Float16* p) {
    Op32 u;
    u.h[0] = *(const v8h*)(p);
    u.h[1] = *(const v8h*)(p + 16);
    return u.v;
}

__global__ __launch_bounds__(256)
void fa2_causal_f16wmma(const float* __restrict__ Q,
                        const float* __restrict__ K,
                        const float* __restrict__ V,
                        float* __restrict__ O) {
    constexpr int S = 2048;
    constexpr int D = 64;
    constexpr float SCALE = 0.125f;          // 1/sqrt(64), folded into Q (exact)

    const int tid  = threadIdx.x;
    const int lane = tid & 31;
    const int wave = tid >> 5;               // 0..7
    const int bh   = blockIdx.y;             // 0..B*H-1
    const int qb   = blockIdx.x;             // query block (128 rows)
    const int q0   = qb * 128 + wave * 16;   // this wave's first query row

    const float* qp = Q + (size_t)bh * S * D;
    const float* kp = K + (size_t)bh * S * D;
    const float* vp = V + (size_t)bh * S * D;
    float*       op = O + (size_t)bh * S * D;

    const int row = lane & 15;               // A: M row | B/C: N col
    const int hi  = lane >> 4;               // half-wave select
    const int kb  = hi * 8;                  // K base within a 32-chunk

    // LDS: double-buffered K tile (2x8K) + V^T tile (2x8K, key axis permuted)
    // + per-wave P tiles 16x64 f16 (16K). Total 48 KB.
    __shared__ _Float16 ldsK[2][64 * 64];    // [buf][key][d]
    __shared__ _Float16 ldsVT[2][64 * 64];   // [buf][d][perm(key)]
    __shared__ _Float16 ldsP[8][16 * 64];    // per-wave, key axis permuted
    _Float16* myP = ldsP[wave];

    // ---- Q tile: 16x64 as two f16 A operands, pre-scaled by 1/sqrt(d) ----
    v16h aq0, aq1;
    {
        const float* qr = qp + (size_t)(q0 + row) * D;
#pragma unroll
        for (int c = 0; c < 2; ++c) {
            v4f f0 = *(const v4f*)(qr + c * 32 + kb);
            v4f f1 = *(const v4f*)(qr + c * 32 + kb + 4);
            v4f f2 = *(const v4f*)(qr + c * 32 + kb + 16);
            v4f f3 = *(const v4f*)(qr + c * 32 + kb + 20);
            Op32 u;
#pragma unroll
            for (int e = 0; e < 4; ++e) {
                u.v[e]      = (_Float16)(f0[e] * SCALE);
                u.v[4 + e]  = (_Float16)(f1[e] * SCALE);
                u.v[8 + e]  = (_Float16)(f2[e] * SCALE);
                u.v[12 + e] = (_Float16)(f3[e] * SCALE);
            }
            if (c) aq1 = u.v; else aq0 = u.v;
        }
    }

    // Accumulators: 16x64 f32 as four 16x16 C tiles + softmax state.
    v8f o0 = {}, o1 = {}, o2 = {}, o3 = {};
    float mr[8], lr[8];
#pragma unroll
    for (int r = 0; r < 8; ++r) { mr[r] = -__builtin_inff(); lr[r] = 0.0f; }

    // Cooperative staging role: key row sr (0..63), 16 d values at sc.
    const int sr = tid >> 2;                 // 0..63
    const int sc = (tid & 3) * 16;           // 0,16,32,48
    // Permuted column for V^T matching packed-P layout:
    // key n -> (n>>5)*32 + 2*(n&15) + ((n>>4)&1)
    const int vcol = ((sr >> 5) << 5) + 2 * (sr & 15) + ((sr >> 4) & 1);

    const int jEnd = (qb + 1) * 128;         // uniform across the block

    for (int j0 = 0; j0 < jEnd; j0 += 64) {
        const int ib = (j0 >> 6) & 1;        // LDS buffer select
        _Float16* bK = ldsK[ib];
        _Float16* bV = ldsVT[ib];

        // ---- stage K (row-major) and V (transposed, permuted keys) ----
        {
            const float* ks = kp + (size_t)(j0 + sr) * D + sc;
            v4f f0 = *(const v4f*)(ks);
            v4f f1 = *(const v4f*)(ks + 4);
            v4f f2 = *(const v4f*)(ks + 8);
            v4f f3 = *(const v4f*)(ks + 12);
            Op32 u;
#pragma unroll
            for (int e = 0; e < 4; ++e) {
                u.v[e]      = (_Float16)f0[e];
                u.v[4 + e]  = (_Float16)f1[e];
                u.v[8 + e]  = (_Float16)f2[e];
                u.v[12 + e] = (_Float16)f3[e];
            }
            *(v8h*)(&bK[sr * 64 + sc])     = u.h[0];
            *(v8h*)(&bK[sr * 64 + sc + 8]) = u.h[1];

            const float* vs = vp + (size_t)(j0 + sr) * D + sc;
            f0 = *(const v4f*)(vs);
            f1 = *(const v4f*)(vs + 4);
            f2 = *(const v4f*)(vs + 8);
            f3 = *(const v4f*)(vs + 12);
#pragma unroll
            for (int e = 0; e < 4; ++e) {
                bV[(sc + e)      * 64 + vcol] = (_Float16)f0[e];
                bV[(sc + 4 + e)  * 64 + vcol] = (_Float16)f1[e];
                bV[(sc + 8 + e)  * 64 + vcol] = (_Float16)f2[e];
                bV[(sc + 12 + e) * 64 + vcol] = (_Float16)f3[e];
            }
        }
        if (j0 + 64 < jEnd) {                // prefetch next key block
            __builtin_prefetch(kp + (size_t)(j0 + 64 + sr) * D + sc, 0, 3);
            __builtin_prefetch(vp + (size_t)(j0 + 64 + sr) * D + sc, 0, 3);
        }
        // Single barrier per iteration: orders stage(i) before compute(i) and
        // (transitively) compute(i) before stage(i+2)'s buffer reuse.
        __syncthreads();

        if (j0 <= q0 + 15) {                 // causal: this wave needs the block
            // -- S = Q*K^T : four 16-key tiles, two pipelined B buffers --
            v8f s0 = {}, s1 = {}, s2 = {}, s3 = {};
            {
                v16h bk0 = lds_operand(&bK[(0  + row) * 64 + kb]);
                v16h bk1 = lds_operand(&bK[(0  + row) * 64 + 32 + kb]);
                s0 = WMMA_F16(aq0, bk0, s0);
                bk0 = lds_operand(&bK[(16 + row) * 64 + kb]);
                s0 = WMMA_F16(aq1, bk1, s0);
                bk1 = lds_operand(&bK[(16 + row) * 64 + 32 + kb]);
                s1 = WMMA_F16(aq0, bk0, s1);
                bk0 = lds_operand(&bK[(32 + row) * 64 + kb]);
                s1 = WMMA_F16(aq1, bk1, s1);
                bk1 = lds_operand(&bK[(32 + row) * 64 + 32 + kb]);
                s2 = WMMA_F16(aq0, bk0, s2);
                bk0 = lds_operand(&bK[(48 + row) * 64 + kb]);
                s2 = WMMA_F16(aq1, bk1, s2);
                bk1 = lds_operand(&bK[(48 + row) * 64 + 32 + kb]);
                s3 = WMMA_F16(aq0, bk0, s3);
                s3 = WMMA_F16(aq1, bk1, s3);
            }

            if (j0 + 63 > q0) {              // diagonal block: causal mask
                const int jc0 = j0 + row, jc1 = jc0 + 16, jc2 = jc0 + 32, jc3 = jc0 + 48;
#pragma unroll
                for (int r = 0; r < 8; ++r) {
                    const int qrow = q0 + r + 8 * hi;
                    if (jc0 > qrow) s0[r] = -__builtin_inff();
                    if (jc1 > qrow) s1[r] = -__builtin_inff();
                    if (jc2 > qrow) s2[r] = -__builtin_inff();
                    if (jc3 > qrow) s3[r] = -__builtin_inff();
                }
            }

            // -- online softmax over 64 keys; packed-P stash into LDS --
#pragma unroll
            for (int r = 0; r < 8; ++r) {
                float mx = fmaxf(fmaxf(s0[r], s1[r]), fmaxf(s2[r], s3[r]));
                mx = rowmax16(mx);
                float mn = fmaxf(mr[r], mx);
                float alpha = __expf(mr[r] - mn);
                float p0 = __expf(s0[r] - mn);
                float p1 = __expf(s1[r] - mn);
                float p2 = __expf(s2[r] - mn);
                float p3 = __expf(s3[r] - mn);
                float rs = rowsum16((p0 + p1) + (p2 + p3));
                lr[r] = lr[r] * alpha + rs;
                mr[r] = mn;
                o0[r] *= alpha; o1[r] *= alpha; o2[r] *= alpha; o3[r] *= alpha;
                // packed dword stores; key axis permutation matches ldsVT
                _Float16* pp = myP + (r + 8 * hi) * 64;
                *(v2h*)(pp + 2 * row)      = (v2h){(_Float16)p0, (_Float16)p1};
                *(v2h*)(pp + 32 + 2 * row) = (v2h){(_Float16)p2, (_Float16)p3};
            }

            // -- O += P*V : A = P (two 32-key chunks), pipelined B buffers --
            v16h pa0 = lds_operand(myP + row * 64 + kb);
            v16h pa1 = lds_operand(myP + row * 64 + 32 + kb);
            {
                v16h bv0 = lds_operand(&bV[(0  + row) * 64 + kb]);
                v16h bv1 = lds_operand(&bV[(0  + row) * 64 + 32 + kb]);
                o0 = WMMA_F16(pa0, bv0, o0);
                bv0 = lds_operand(&bV[(16 + row) * 64 + kb]);
                o0 = WMMA_F16(pa1, bv1, o0);
                bv1 = lds_operand(&bV[(16 + row) * 64 + 32 + kb]);
                o1 = WMMA_F16(pa0, bv0, o1);
                bv0 = lds_operand(&bV[(32 + row) * 64 + kb]);
                o1 = WMMA_F16(pa1, bv1, o1);
                bv1 = lds_operand(&bV[(32 + row) * 64 + 32 + kb]);
                o2 = WMMA_F16(pa0, bv0, o2);
                bv0 = lds_operand(&bV[(48 + row) * 64 + kb]);
                o2 = WMMA_F16(pa1, bv1, o2);
                bv1 = lds_operand(&bV[(48 + row) * 64 + 32 + kb]);
                o3 = WMMA_F16(pa0, bv0, o3);
                o3 = WMMA_F16(pa1, bv1, o3);
            }
        }
        // No trailing barrier: next iteration writes the other LDS buffer.
    }

    // ---- normalize by l and store fp32 output ----
#pragma unroll
    for (int r = 0; r < 8; ++r) {
        const float inv = 1.0f / lr[r];
        float* orow = op + (size_t)(q0 + r + 8 * hi) * D + row;
        orow[0]  = o0[r] * inv;
        orow[16] = o1[r] * inv;
        orow[32] = o2[r] * inv;
        orow[48] = o3[r] * inv;
    }
}

extern "C" void kernel_launch(void* const* d_in, const int* in_sizes, int n_in,
                              void* d_out, int out_size, void* d_ws, size_t ws_size,
                              hipStream_t stream) {
    (void)in_sizes; (void)n_in; (void)out_size; (void)d_ws; (void)ws_size;
    const float* q = (const float*)d_in[0];
    const float* k = (const float*)d_in[1];
    const float* v = (const float*)d_in[2];
    // d_in[3] is the causal mask; causality is computed analytically.
    float* out = (float*)d_out;

    constexpr int B = 4, H = 16, S = 2048;
    dim3 grid(S / 128, B * H);   // (query blocks, batch*heads)
    dim3 block(256);             // 8 waves of 32
    fa2_causal_f16wmma<<<grid, block, 0, stream>>>(q, k, v, out);
}
